// MMN_49933289783472
// MI455X (gfx1250) — matrix operand order repaired and checked by reading, not compiled, not run on hardware
//
#include <hip/hip_runtime.h>

// ---------------------------------------------------------------------------
// Problem constants (from reference): B=2, h=w=60, hw=3600
// corr = mean(S4, S3);  attn = softmax(20*corr);  att_fq = attn @ f_s^T
// fq = l2n(f_q) + 0.5 * l2n(att_fq)
// Fold 20*0.5 = 10 into the packed (normalized) query features so a single
// K=3072 bf16 GEMM produces the temperature-scaled mean corr directly.
// ---------------------------------------------------------------------------
#define NB 2
#define HW 3600
#define C4F 2048
#define C3F 1024
#define CT  3072   // C4F + C3F
#define CV  256

typedef __attribute__((ext_vector_type(8)))  __bf16 v8bf;
typedef __attribute__((ext_vector_type(16))) __bf16 v16bf;
typedef __attribute__((ext_vector_type(8)))  float  v8f;
typedef __attribute__((ext_vector_type(4)))  unsigned v4u;
typedef __attribute__((ext_vector_type(4)))  int    v4i;
typedef __attribute__((ext_vector_type(8)))  int    v8i;

__device__ __forceinline__ __bf16 to_bf16(float f) {
  unsigned u = __float_as_uint(f);
  u += 0x7FFFu + ((u >> 16) & 1u);     // round-to-nearest-even
  unsigned short h = (unsigned short)(u >> 16);
  __bf16 r;
  __builtin_memcpy(&r, &h, sizeof(h));
  return r;
}

// ---------------------------------------------------------------------------
// Pass 1: per-pixel channel L2 inv-norms for fq4, fq3, fs4, fs3.
// inv[j][b][q], j = 0:q4 1:q3 2:s4 3:s3. Coalesced over q.
// ---------------------------------------------------------------------------
__global__ __launch_bounds__(256) void norms_kernel(
    const float* __restrict__ fq4, const float* __restrict__ fq3,
    const float* __restrict__ fs4, const float* __restrict__ fs3,
    float* __restrict__ inv)
{
  int idx = blockIdx.x * 256 + threadIdx.x;
  if (idx >= NB * HW) return;
  int b = idx / HW, q = idx - b * HW;
  float a = 0.f, c = 0.f, d = 0.f, e = 0.f;
  const float* p4q = fq4 + (size_t)b * C4F * HW + q;
  const float* p4s = fs4 + (size_t)b * C4F * HW + q;
  for (int k = 0; k < C4F; ++k) {
    float x = p4q[(size_t)k * HW]; a += x * x;
    float y = p4s[(size_t)k * HW]; d += y * y;
  }
  const float* p3q = fq3 + (size_t)b * C3F * HW + q;
  const float* p3s = fs3 + (size_t)b * C3F * HW + q;
  for (int k = 0; k < C3F; ++k) {
    float x = p3q[(size_t)k * HW]; c += x * x;
    float y = p3s[(size_t)k * HW]; e += y * y;
  }
  inv[0 * NB * HW + idx] = 1.f / fmaxf(sqrtf(a), 1e-12f);
  inv[1 * NB * HW + idx] = 1.f / fmaxf(sqrtf(c), 1e-12f);
  inv[2 * NB * HW + idx] = 1.f / fmaxf(sqrtf(d), 1e-12f);
  inv[3 * NB * HW + idx] = 1.f / fmaxf(sqrtf(e), 1e-12f);
}

// ---------------------------------------------------------------------------
// Pass 2a: Qp[b][q][k] = bf16(10 * fq{4,3}[b][c][q] * inv_q)  (LDS transpose)
// Row-major [q][3072] => WMMA A-fragments are contiguous 16B chunks.
// ---------------------------------------------------------------------------
__global__ __launch_bounds__(256) void packQ_kernel(
    const float* __restrict__ fq4, const float* __restrict__ fq3,
    const float* __restrict__ inv, __bf16* __restrict__ Qp)
{
  __shared__ float tile[32][33];
  int b = blockIdx.z, c0 = blockIdx.x * 32, q0 = blockIdx.y * 32;
  int tx = threadIdx.x, ty = threadIdx.y;
  #pragma unroll
  for (int i = 0; i < 4; ++i) {
    int c = c0 + ty + i * 8, q = q0 + tx;
    float v = 0.f;
    if (q < HW) {
      if (c < C4F) v = fq4[((size_t)(b * C4F + c)) * HW + q] * inv[0 * NB * HW + b * HW + q];
      else         v = fq3[((size_t)(b * C3F + (c - C4F))) * HW + q] * inv[1 * NB * HW + b * HW + q];
    }
    tile[ty + i * 8][tx] = v * 10.0f;   // fold temp*mean = 20*0.5
  }
  __syncthreads();
  #pragma unroll
  for (int i = 0; i < 4; ++i) {
    int q = q0 + ty + i * 8, c = c0 + tx;
    if (q < HW) Qp[((size_t)b * HW + q) * CT + c] = to_bf16(tile[tx][ty + i * 8]);
  }
}

// ---------------------------------------------------------------------------
// Pass 2b: Sp[b][k][s] = bf16(fs{4,3}[b][c][s] * inv_s)  (layout preserving)
// Channel-major => each WMMA B-fragment k row is 32B contiguous across 16 lanes (lane = N = s).
// ---------------------------------------------------------------------------
__global__ __launch_bounds__(256) void packS_kernel(
    const float* __restrict__ fs4, const float* __restrict__ fs3,
    const float* __restrict__ inv, __bf16* __restrict__ Sp)
{
  size_t idx = (size_t)blockIdx.x * 256 + threadIdx.x;
  if (idx >= (size_t)NB * CT * HW) return;
  int b = (int)(idx / ((size_t)CT * HW));
  size_t r = idx - (size_t)b * CT * HW;
  int k = (int)(r / HW), s = (int)(r - (size_t)k * HW);
  float v;
  if (k < C4F) v = fs4[((size_t)(b * C4F + k)) * HW + s] * inv[2 * NB * HW + b * HW + s];
  else         v = fs3[((size_t)(b * C3F + (k - C4F))) * HW + s] * inv[3 * NB * HW + b * HW + s];
  Sp[idx] = to_bf16(v);
}

// ---------------------------------------------------------------------------
// Pass 2c: Vp[b][s][c] = bf16(f_s[b][c][s])  (transpose for attn@V B-frags)
// ---------------------------------------------------------------------------
__global__ __launch_bounds__(256) void packV_kernel(
    const float* __restrict__ f_s, __bf16* __restrict__ Vp)
{
  __shared__ float tile[32][33];
  int b = blockIdx.z, c0 = blockIdx.x * 32, q0 = blockIdx.y * 32;
  int tx = threadIdx.x, ty = threadIdx.y;
  #pragma unroll
  for (int i = 0; i < 4; ++i) {
    int c = c0 + ty + i * 8, q = q0 + tx;
    float v = 0.f;
    if (q < HW) v = f_s[((size_t)(b * CV + c)) * HW + q];
    tile[ty + i * 8][tx] = v;
  }
  __syncthreads();
  #pragma unroll
  for (int i = 0; i < 4; ++i) {
    int q = q0 + ty + i * 8, c = c0 + tx;
    if (q < HW) Vp[((size_t)b * HW + q) * CV + c] = to_bf16(tile[tx][ty + i * 8]);
  }
}

// ---------------------------------------------------------------------------
// Fused flash-style kernel: one block (8 waves) per 16 query rows.
// The 16x3072 bf16 Q tile (96 KB) is DMA'd into LDS once per block by the
// Tensor Data Mover (tensor_load_to_lds + s_wait_tensorcnt), so the hot loop
// reads A-fragments via ds_load_b128 instead of re-streaming Q from L2 every
// s-step. Per 128-wide s step: each wave computes one 16x16 corr tile via 96
// v_wmma_f32_16x16x32_bf16 (K=3072), online softmax across waves via LDS,
// exp tile requantized to bf16, then each wave accumulates its 32-channel
// slice of O with 8 more WMMAs. corr volume is never materialized.
// ---------------------------------------------------------------------------
__global__ __launch_bounds__(256, 1) void fused_attn_kernel(
    const __bf16* __restrict__ Qp, const __bf16* __restrict__ Sp,
    const __bf16* __restrict__ Vp, const float* __restrict__ f_q,
    float* __restrict__ out_fq, float* __restrict__ out_att)
{
  __shared__ __align__(32) __bf16 qtile[16][CT];    // 96 KB, TDM-filled
  __shared__ __align__(32) __bf16 ptile[16][128];   // exp(corr) tile, bf16
  __shared__ float otile[16][CV];
  __shared__ float wmax[8][16];
  __shared__ float wsum[8][16];
  __shared__ float rowM[16], rowL[16], rowA[16];
  __shared__ float inv_attn[16], inv_fqn[16];

  const int tid  = threadIdx.x;
  const int wid  = tid >> 5;
  const int lane = tid & 31;
  const int half = lane >> 4;      // ISA 7.12.2: lanes 16-31 hold M+8 / K-hi
  const int mrow = lane & 15;
  const int b  = blockIdx.y;
  const int q0 = blockIdx.x * 16;

  if (tid < 16) { rowM[tid] = -1e30f; rowL[tid] = 0.f; }

  // ---- TDM: DMA the 16x3072 bf16 Q tile into LDS (wave 0 issues) ---------
  if (wid == 0) {
    // D# group0: count=1 | lds_addr | global_addr(57b) | type=2
    unsigned lds_addr = (unsigned)(unsigned long long)&qtile[0][0]; // LDS ap.
    unsigned long long ga =
        (unsigned long long)(Qp + ((size_t)(b * HW + q0)) * CT);
    v4u g0;
    g0[0] = 1u;                                  // count=1, user mode
    g0[1] = lds_addr;                            // D#.lds_addr
    g0[2] = (unsigned)ga;                        // global_addr[31:0]
    g0[3] = (unsigned)(ga >> 32) | (2u << 30);   // global_addr[56:32]|type=2
    // D# group1: data_size=2B, tensor 3600x3072, tile 16x3072, stride 3072
    v8i g1;
    g1[0] = 0x00010000;          // [17:16] data_size=1 (2 bytes), mask=0
    g1[1] = (int)(3072u << 16);  // [63:48]  tensor_dim0 lo16
    g1[2] = (int)(3600u << 16);  // [95:80]  tensor_dim1 lo16 (dim0 hi16=0)
    g1[3] = (int)(3072u << 16);  // [127:112] tile_dim0    (dim1 hi16=0)
    g1[4] = 16;                  // [143:128] tile_dim1 ; tile_dim2=0
    g1[5] = 3072;                // [191:160] tensor_dim0_stride lo32
    g1[6] = 0;                   // stride hi / dim1_stride lo
    g1[7] = 0;
    v4i z4 = {0, 0, 0, 0};
#if __clang_major__ >= 23
    v8i z8 = {0, 0, 0, 0, 0, 0, 0, 0};
    __builtin_amdgcn_tensor_load_to_lds(g0, g1, z4, z4, z8, 0);
#else
    __builtin_amdgcn_tensor_load_to_lds(g0, g1, z4, z4, 0);
#endif
    __builtin_amdgcn_s_wait_tensorcnt(0);        // TENSORcnt -> 0
  }
  __syncthreads();                               // qtile visible to all waves

  v8f oacc0 = {}; v8f oacc1 = {};   // O slices: channels [32w,32w+16), [32w+16,32w+32)

  // A-fragment base in LDS: row M = mrow, K chunks at k+8*half, k+16+8*half.
  const __bf16* qlo = &qtile[mrow][half * 8];
  const __bf16* qhi = &qtile[mrow][16 + half * 8];

  for (int s0 = 0; s0 < HW; s0 += 128) {
    const int sw = s0 + wid * 16;          // this wave's s tile
    const bool active = sw < HW;           // HW%16==0 -> tiles all-or-nothing
    v8f p = {};
    if (active) {
      // B-fragment: column N = mrow, 16 K values (table order, stride HW) per lane.
      const __bf16* bp = Sp + ((size_t)(b * CT)) * HW + sw + mrow;
      #pragma unroll 4
      for (int k0 = 0; k0 < CT; k0 += 32) {
        v8bf alo = *(const v8bf*)(qlo + k0);     // ds_load_b128
        v8bf ahi = *(const v8bf*)(qhi + k0);     // ds_load_b128
        v16bf A = __builtin_shufflevector(alo, ahi,
                    0,1,2,3,4,5,6,7,8,9,10,11,12,13,14,15);
        v16bf Bm;
        for (int j = 0; j < 16; ++j) Bm[j] = bp[(size_t)(j < 8 ? j + 8 * half : 8 + j + 8 * half) * HW];
        bp += (size_t)32 * HW;
        __builtin_prefetch(bp, 0, 1);            // global_prefetch_b8
        p = __builtin_amdgcn_wmma_f32_16x16x32_bf16(
                false, A, false, Bm, (short)0, p, false, false);
      }
    }
    // Per-row max inside this wave's 16x16 tile (row m = r + 8*half).
    #pragma unroll
    for (int r = 0; r < 8; ++r) {
      float v = active ? p[r] : -1e30f;
      v = fmaxf(v, __shfl_xor(v, 8, 32));
      v = fmaxf(v, __shfl_xor(v, 4, 32));
      v = fmaxf(v, __shfl_xor(v, 2, 32));
      v = fmaxf(v, __shfl_xor(v, 1, 32));
      if (mrow == 0) wmax[wid][r + 8 * half] = v;
    }
    __syncthreads();
    if (tid < 16) {                         // combine across waves
      float m = rowM[tid];
      #pragma unroll
      for (int w = 0; w < 8; ++w) m = fmaxf(m, wmax[w][tid]);
      rowA[tid] = __expf(rowM[tid] - m);
      rowM[tid] = m;
    }
    __syncthreads();
    // exp, row sums, requantize exp tile to bf16 into LDS.
    #pragma unroll
    for (int r = 0; r < 8; ++r) {
      int row = r + 8 * half;
      float e = active ? __expf(p[r] - rowM[row]) : 0.f;
      ptile[row][wid * 16 + mrow] = to_bf16(e);
      float s = e;
      s += __shfl_xor(s, 8, 32);
      s += __shfl_xor(s, 4, 32);
      s += __shfl_xor(s, 2, 32);
      s += __shfl_xor(s, 1, 32);
      if (mrow == 0) wsum[wid][row] = s;
    }
    __syncthreads();
    if (tid < 16) {
      float s = 0.f;
      #pragma unroll
      for (int w = 0; w < 8; ++w) s += wsum[w][tid];
      rowL[tid] = rowL[tid] * rowA[tid] + s;
    }
    // Rescale O by alpha (per query row), then O += exp(P) @ V slice.
    #pragma unroll
    for (int r = 0; r < 8; ++r) {
      float al = rowA[r + 8 * half];
      oacc0[r] *= al; oacc1[r] *= al;
    }
    #pragma unroll
    for (int k0 = 0; k0 < 128; k0 += 32) {
      v8bf plo = *(const v8bf*)&ptile[mrow][k0 + half * 8];
      v8bf phi = *(const v8bf*)&ptile[mrow][k0 + 16 + half * 8];
      v16bf A = __builtin_shufflevector(plo, phi,
                  0,1,2,3,4,5,6,7,8,9,10,11,12,13,14,15);
      v16bf B0, B1;
      for (int j = 0; j < 16; ++j) {
        int sidx = s0 + k0 + (j < 8 ? j + 8 * half : 8 + j + 8 * half);
        if (sidx >= HW) sidx = HW - 1;       // padded cols multiply by exp=0
        const __bf16* vb = Vp + ((size_t)(b * HW + sidx)) * CV + wid * 32 + mrow;
        B0[j] = vb[0];
        B1[j] = vb[16];
      }
      oacc0 = __builtin_amdgcn_wmma_f32_16x16x32_bf16(
                  false, A, false, B0, (short)0, oacc0, false, false);
      oacc1 = __builtin_amdgcn_wmma_f32_16x16x32_bf16(
                  false, A, false, B1, (short)0, oacc1, false, false);
    }
    __syncthreads();
  }

  // Finalize: divide by l, stage O, compute both channel L2 norms, emit.
  #pragma unroll
  for (int r = 0; r < 8; ++r) {
    int row = r + 8 * half;
    float invl = 1.0f / rowL[row];
    otile[row][wid * 32 + mrow]      = oacc0[r] * invl;
    otile[row][wid * 32 + 16 + mrow] = oacc1[r] * invl;
  }
  __syncthreads();
  if (tid < 16) {
    float sa = 0.f, sq = 0.f;
    const float* fqp = f_q + (size_t)b * CV * HW + (q0 + tid);
    for (int c = 0; c < CV; ++c) {
      float av = otile[tid][c];      sa += av * av;
      float qv = fqp[(size_t)c * HW]; sq += qv * qv;
    }
    inv_attn[tid] = 1.f / fmaxf(sqrtf(sa), 1e-12f);
    inv_fqn[tid]  = 1.f / fmaxf(sqrtf(sq), 1e-12f);
  }
  __syncthreads();
  {
    int ql = tid & 15;
    int cb = (tid >> 4) * 16;
    size_t q = (size_t)q0 + ql;
    #pragma unroll
    for (int i = 0; i < 16; ++i) {
      int c = cb + i;
      size_t off = ((size_t)b * CV + c) * HW + q;
      float av = otile[ql][c];
      out_att[off] = av;
      out_fq[off]  = f_q[off] * inv_fqn[ql] + 0.5f * av * inv_attn[ql];
    }
  }
}

// ---------------------------------------------------------------------------
extern "C" void kernel_launch(void* const* d_in, const int* in_sizes, int n_in,
                              void* d_out, int out_size, void* d_ws, size_t ws_size,
                              hipStream_t stream) {
  (void)in_sizes; (void)n_in; (void)out_size; (void)ws_size;
  const float* fq3 = (const float*)d_in[0];
  const float* fs3 = (const float*)d_in[1];
  const float* fq4 = (const float*)d_in[2];
  const float* fs4 = (const float*)d_in[3];
  const float* f_q = (const float*)d_in[4];
  const float* f_s = (const float*)d_in[5];

  float* out_fq  = (float*)d_out;
  float* out_att = out_fq + (size_t)NB * CV * HW;

  char* w = (char*)d_ws;
  const size_t invBytes = (size_t)4 * NB * HW * sizeof(float);   // 115200
  const size_t qpBytes  = (size_t)NB * HW * CT * sizeof(__bf16); // ~44.2 MB
  float*  inv = (float*)w;
  __bf16* Qp  = (__bf16*)(w + invBytes);
  __bf16* Sp  = (__bf16*)(w + invBytes + qpBytes);
  __bf16* Vp  = (__bf16*)(w + invBytes + 2 * qpBytes);

  norms_kernel<<<(NB * HW + 255) / 256, 256, 0, stream>>>(fq4, fq3, fs4, fs3, inv);
  packQ_kernel<<<dim3(CT / 32, (HW + 31) / 32, NB), dim3(32, 8), 0, stream>>>(fq4, fq3, inv, Qp);
  packS_kernel<<<(unsigned)(((size_t)NB * CT * HW + 255) / 256), 256, 0, stream>>>(fs4, fs3, inv, Sp);
  packV_kernel<<<dim3(CV / 32, (HW + 31) / 32, NB), dim3(32, 8), 0, stream>>>(f_s, Vp);
  fused_attn_kernel<<<dim3(HW / 16, NB), 256, 0, stream>>>(Qp, Sp, Vp, f_q, out_fq, out_att);
}
